// MultiHeadAttention_300647711335
// MI455X (gfx1250) — matrix operand order repaired and checked by reading, not compiled, not run on hardware
//
#include <hip/hip_runtime.h>
#include <hip/hip_fp16.h>

#define NB    4
#define SEQ   2048
#define NH    16
#define NE    1024
#define HD    64
#define MROWS (NB * SEQ)   // 8192

typedef _Float16 v8h  __attribute__((ext_vector_type(8)));
typedef _Float16 v16h __attribute__((ext_vector_type(16)));
typedef float    v8f  __attribute__((ext_vector_type(8)));
typedef unsigned int v4u __attribute__((ext_vector_type(4)));
typedef int      v8i  __attribute__((ext_vector_type(8)));
typedef int      v4i  __attribute__((ext_vector_type(4)));

// Use the Tensor Data Mover when the builtin exists (this toolchain: 6-arg form).
#if defined(__HIP_DEVICE_COMPILE__) && defined(__has_builtin)
#if __has_builtin(__builtin_amdgcn_tensor_load_to_lds) && __has_builtin(__builtin_amdgcn_s_wait_tensorcnt)
#define USE_TDM 1
#endif
#endif
#ifndef USE_TDM
#define USE_TDM 0
#endif

// ---------------- WMMA operand helpers (wave32, 16x16x32 f16) ----------------

__device__ inline v16h cat8(v8h lo, v8h hi) {
  return __builtin_shufflevector(lo, hi, 0,1,2,3,4,5,6,7,8,9,10,11,12,13,14,15);
}

// A operand: 16x32 f16 tile from row-major [*, lda] matrix.
// lane<16: row=lane,   K chunks {k0+0..7, k0+16..23}
// lane>=16: row=lane-16, K chunks {k0+8..15, k0+24..31}
__device__ inline v16h load_a16(const _Float16* base, int row0, int k0, int lda, int lane) {
  int m    = lane & 15;
  int koff = (lane < 16) ? 0 : 8;
  const _Float16* p = base + (size_t)(row0 + m) * lda + k0 + koff;
  v8h lo = *(const v8h*)p;
  v8h hi = *(const v8h*)(p + 16);
  return cat8(lo, hi);
}

// B operand: 32x16 f16 tile; source stored "transposed" row-major [col][K] with
// leading dim ldk.  lane<16: col=lane, K chunks {k0+0..7, k0+16..23}; lane>=16: col=lane-16, K chunks {k0+8..15, k0+24..31}
__device__ inline v16h load_b16(const _Float16* baseT, int col0, int k0, int ldk, int lane) {
  int n    = lane & 15;
  int koff = (lane < 16) ? 0 : 8;
  const _Float16* p = baseT + (size_t)(col0 + n) * ldk + k0 + koff;
  v8h lo = *(const v8h*)p;
  v8h hi = *(const v8h*)(p + 16);
  return cat8(lo, hi);
}

__device__ inline v8f wmma_f16(v16h a, v16h b, v8f c) {
  return __builtin_amdgcn_wmma_f32_16x16x32_f16(false, a, false, b, (short)0, c, false, false);
}

// ---------------- elementwise / layout kernels ----------------

__global__ void k_cvt_f16(const float* __restrict__ in, _Float16* __restrict__ out, int n) {
  int i = blockIdx.x * blockDim.x + threadIdx.x;
  if (i < n) out[i] = (_Float16)in[i];
}

// W [K=1024][N=1024] f32  ->  Wt f16 [N][K]
__global__ void k_w_transpose(const float* __restrict__ W, _Float16* __restrict__ Wt) {
  int i = blockIdx.x * blockDim.x + threadIdx.x;
  int n = i >> 10, k = i & (NE - 1);
  Wt[(size_t)n * NE + k] = (_Float16)W[(size_t)k * NE + n];
}

// Y [8192][1024] f16 -> out [b][h][n][64] f16, scaled
__global__ void k_reshape_qk(const _Float16* __restrict__ Y, _Float16* __restrict__ out, float scale) {
  int i = blockIdx.x * blockDim.x + threadIdx.x;
  int col = i & (NE - 1), row = i >> 10;
  int b = row >> 11, n = row & (SEQ - 1);
  int h = col >> 6,  d = col & (HD - 1);
  out[((((size_t)b * NH + h) * SEQ + n) << 6) + d] = (_Float16)((float)Y[i] * scale);
}

// Y [8192][1024] f16 -> out [b][h][64][n] f16  (V pre-transposed per head)
__global__ void k_reshape_vt(const _Float16* __restrict__ Y, _Float16* __restrict__ out) {
  int i = blockIdx.x * blockDim.x + threadIdx.x;
  int col = i & (NE - 1), row = i >> 10;
  int b = row >> 11, n = row & (SEQ - 1);
  int h = col >> 6,  d = col & (HD - 1);
  out[(((size_t)b * NH + h) * HD + d) * SEQ + n] = Y[i];
}

// ---------------- GEMM: C[M,N] = A[M,K] * Bt[N,K]^T (+bias, f32 out) ----------------
// block = 128 threads = 4 waves in a 2x2 grid; each wave owns a 32x32 f32 tile.

__global__ __launch_bounds__(128) void k_gemm(const _Float16* __restrict__ A,
                                              const _Float16* __restrict__ Bt,
                                              _Float16* __restrict__ C16,
                                              float* __restrict__ C32,
                                              const float* __restrict__ bias,
                                              int M, int N, int K) {
  int lane = threadIdx.x & 31;
  int w    = threadIdx.x >> 5;
  int r0 = blockIdx.x * 64 + (w >> 1) * 32;
  int c0 = blockIdx.y * 64 + (w & 1) * 32;

  v8f acc[2][2] = {};
  for (int k0 = 0; k0 < K; k0 += 32) {
    v16h a0 = load_a16(A, r0,      k0, K, lane);
    v16h a1 = load_a16(A, r0 + 16, k0, K, lane);
    v16h b0 = load_b16(Bt, c0,      k0, K, lane);
    v16h b1 = load_b16(Bt, c0 + 16, k0, K, lane);
    acc[0][0] = wmma_f16(a0, b0, acc[0][0]);
    acc[0][1] = wmma_f16(a0, b1, acc[0][1]);
    acc[1][0] = wmma_f16(a1, b0, acc[1][0]);
    acc[1][1] = wmma_f16(a1, b1, acc[1][1]);
  }

  int coff = (lane < 16) ? 0 : 8;
  int nl   = lane & 15;
  for (int i = 0; i < 2; i++)
    for (int j = 0; j < 2; j++)
      for (int r = 0; r < 8; r++) {
        size_t row = (size_t)(r0 + 16 * i + r + coff);
        int    col = c0 + 16 * j + nl;
        if (C32) C32[row * N + col] = acc[i][j][r] + bias[col];
        else     C16[row * N + col] = (_Float16)acc[i][j][r];
      }
}

// ---------------- Flash attention ----------------
// grid (SEQ/64, NH, NB); block = 128 = 4 waves; wave w owns 16 query rows.

__global__ __launch_bounds__(128) void k_flash(const _Float16* __restrict__ Q,
                                               const _Float16* __restrict__ Kh,
                                               const _Float16* __restrict__ Vt,
                                               _Float16* __restrict__ ctx) {
  __shared__ alignas(16) _Float16 sK[32 * HD];   // K tile  [32 keys][64 dims]
  __shared__ alignas(16) _Float16 sV[HD * 32];   // Vt tile [64 dims][32 keys]
  __shared__ alignas(16) float    sS[4][16][32]; // per-wave score staging

  int lane = threadIdx.x & 31;
  int w    = threadIdx.x >> 5;
  int tile = blockIdx.x, h = blockIdx.y, b = blockIdx.z;
  int bh   = b * NH + h;

  const _Float16* Qb = Q  + (size_t)bh * SEQ * HD;
  const _Float16* Kb = Kh + (size_t)bh * SEQ * HD;
  const _Float16* Vb = Vt + (size_t)bh * HD * SEQ;

  int q0 = tile * 64 + w * 16;
  v16h aq0 = load_a16(Qb, q0, 0,  HD, lane);
  v16h aq1 = load_a16(Qb, q0, 32, HD, lane);

  v8f   o[4] = {};
  float m_i = -__builtin_inff();
  float l_i = 0.0f;
  int coff = (lane < 16) ? 0 : 8;
  int nl   = lane & 15;

  int kend = (tile + 1) * 64;           // causal: keys <= last query row of tile
  for (int kk = 0; kk < kend; kk += 32) {
    __syncthreads();                    // previous tile fully consumed

#if USE_TDM
    if (w == 0) {
      unsigned ldsK = (unsigned)(size_t)&sK[0];
      unsigned ldsV = (unsigned)(size_t)&sV[0];
      unsigned long long gk = (unsigned long long)(size_t)(Kb + (size_t)kk * HD);
      unsigned long long gv = (unsigned long long)(size_t)(Vb + (size_t)kk);
      // D# group0: count=1, lds_addr, global_addr[56:0], type=2
      v4u g0k = { 1u, ldsK, (unsigned)gk,
                  (unsigned)((gk >> 32) & 0x1ffffffu) | 0x80000000u };
      v4u g0v = { 1u, ldsV, (unsigned)gv,
                  (unsigned)((gv >> 32) & 0x1ffffffu) | 0x80000000u };
      // D# group1: data_size=2B; K tile: dim0=64 stride 64, 64x32 tile
      v8i g1k = { (int)(1u << 16), (int)(64u << 16), (int)(2048u << 16),
                  (int)(64u << 16), 32, 64, 0, 0 };
      // Vt tile: dim0=2048 stride 2048, 32x64 tile
      v8i g1v = { (int)(1u << 16), (int)(2048u << 16), (int)(64u << 16),
                  (int)(32u << 16), 64, 2048, 0, 0 };
      v4i z4 = { 0, 0, 0, 0 };
      v8i z8 = { 0, 0, 0, 0, 0, 0, 0, 0 };
      __builtin_amdgcn_tensor_load_to_lds(g0k, g1k, z4, z4, z8, 0);
      __builtin_amdgcn_tensor_load_to_lds(g0v, g1v, z4, z4, z8, 0);
      __builtin_amdgcn_s_wait_tensorcnt(0);
    }
#else
    {
      int t = threadIdx.x;
      v8h* dK = (v8h*)sK;
      const _Float16* gK = Kb + (size_t)kk * HD;
      dK[t]       = *(const v8h*)(gK + (size_t)t * 8);
      dK[t + 128] = *(const v8h*)(gK + (size_t)(t + 128) * 8);
      v8h* dV = (v8h*)sV;
      for (int i = t; i < 256; i += 128) {
        int d = i >> 2, c = i & 3;
        dV[i] = *(const v8h*)(Vb + (size_t)d * SEQ + kk + c * 8);
      }
    }
#endif
    __syncthreads();

    // S = Q * K^T  (16x32 per wave, f32)
    v8f s0 = {}, s1 = {};
    {
      v16h bk = load_b16(sK, 0, 0, HD, lane);
      s0 = wmma_f16(aq0, bk, s0);
      bk = load_b16(sK, 0, 32, HD, lane);
      s0 = wmma_f16(aq1, bk, s0);
      bk = load_b16(sK, 16, 0, HD, lane);
      s1 = wmma_f16(aq0, bk, s1);
      bk = load_b16(sK, 16, 32, HD, lane);
      s1 = wmma_f16(aq1, bk, s1);
    }

    // causal mask + stage to LDS (C layout -> row-major)
    for (int r = 0; r < 8; r++) {
      int row = q0 + r + coff;
      int c0i = kk + nl, c1i = kk + 16 + nl;
      sS[w][r + coff][nl]      = (c0i <= row) ? s0[r] : -__builtin_inff();
      sS[w][r + coff][16 + nl] = (c1i <= row) ? s1[r] : -__builtin_inff();
    }
    asm volatile("s_wait_dscnt 0" ::: "memory");

    // online softmax in lane-pair row domain:
    // lanes (nl, nl+16) own row nl; lane<16 covers cols {0..7,16..23}, lane>=16 {8..15,24..31}
    int cb = (lane < 16) ? 0 : 8;
    float4 x0 = *(float4*)&sS[w][nl][cb];
    float4 x1 = *(float4*)&sS[w][nl][cb + 4];
    float4 x2 = *(float4*)&sS[w][nl][cb + 16];
    float4 x3 = *(float4*)&sS[w][nl][cb + 20];

    float pm = fmaxf(fmaxf(fmaxf(x0.x, x0.y), fmaxf(x0.z, x0.w)),
                     fmaxf(fmaxf(x1.x, x1.y), fmaxf(x1.z, x1.w)));
    pm = fmaxf(pm, fmaxf(fmaxf(fmaxf(x2.x, x2.y), fmaxf(x2.z, x2.w)),
                         fmaxf(fmaxf(x3.x, x3.y), fmaxf(x3.z, x3.w))));
    pm = fmaxf(pm, __shfl_xor(pm, 16, 32));
    float m_new = fmaxf(m_i, pm);
    float alpha = __expf(m_i - m_new);

    float pf[16];
    pf[0]  = __expf(x0.x - m_new); pf[1]  = __expf(x0.y - m_new);
    pf[2]  = __expf(x0.z - m_new); pf[3]  = __expf(x0.w - m_new);
    pf[4]  = __expf(x1.x - m_new); pf[5]  = __expf(x1.y - m_new);
    pf[6]  = __expf(x1.z - m_new); pf[7]  = __expf(x1.w - m_new);
    pf[8]  = __expf(x2.x - m_new); pf[9]  = __expf(x2.y - m_new);
    pf[10] = __expf(x2.z - m_new); pf[11] = __expf(x2.w - m_new);
    pf[12] = __expf(x3.x - m_new); pf[13] = __expf(x3.y - m_new);
    pf[14] = __expf(x3.z - m_new); pf[15] = __expf(x3.w - m_new);

    float ps = 0.0f;
    for (int i = 0; i < 16; i++) ps += pf[i];
    ps += __shfl_xor(ps, 16, 32);
    l_i = l_i * alpha + ps;
    m_i = m_new;

    // rescale O (broadcast alpha from softmax domain into C-layout domain)
    for (int r = 0; r < 8; r++) {
      float al = __shfl(alpha, r + coff, 32);
      for (int jd = 0; jd < 4; jd++) o[jd][r] *= al;
    }

    // P already sits in the 16-bit A-operand layout
    v16h pv;
    for (int i = 0; i < 16; i++) pv[i] = (_Float16)pf[i];

    // O += P * V   (B from transposed V tile: contiguous per lane)
    for (int jd = 0; jd < 4; jd++) {
      v16h bv = load_b16(sV, 16 * jd, 0, 32, lane);
      o[jd] = wmma_f16(pv, bv, o[jd]);
    }
  }

  // normalize and write ctx[b][q][h*64+d] as f16
  for (int r = 0; r < 8; r++) {
    float li  = __shfl(l_i, r + coff, 32);
    float inv = 1.0f / li;
    size_t row  = (size_t)(q0 + r + coff);
    size_t base = ((size_t)b * SEQ + row) * NE + h * HD;
    for (int jd = 0; jd < 4; jd++)
      ctx[base + 16 * jd + nl] = (_Float16)(o[jd][r] * inv);
  }
}

// ---------------- host launcher ----------------

extern "C" void kernel_launch(void* const* d_in, const int* in_sizes, int n_in,
                              void* d_out, int out_size, void* d_ws, size_t ws_size,
                              hipStream_t stream) {
  const float* x  = (const float*)d_in[0];
  const float* Wq = (const float*)d_in[1];
  const float* Wk = (const float*)d_in[2];
  const float* Wv = (const float*)d_in[3];
  const float* Wo = (const float*)d_in[4];
  const float* bo = (const float*)d_in[5];
  float* out = (float*)d_out;

  char* ws = (char*)d_ws;
  size_t off = 0;
  auto take = [&](size_t bytes) -> void* {
    void* p = ws + off;
    off += (bytes + 255) & ~(size_t)255;
    return p;
  };
  _Float16* xh  = (_Float16*)take((size_t)MROWS * NE * 2);
  _Float16* Wqt = (_Float16*)take((size_t)NE * NE * 2);
  _Float16* Wkt = (_Float16*)take((size_t)NE * NE * 2);
  _Float16* Wvt = (_Float16*)take((size_t)NE * NE * 2);
  _Float16* Wot = (_Float16*)take((size_t)NE * NE * 2);
  _Float16* Y   = (_Float16*)take((size_t)MROWS * NE * 2);
  _Float16* Qh  = (_Float16*)take((size_t)MROWS * NE * 2);
  _Float16* Khb = (_Float16*)take((size_t)MROWS * NE * 2);
  _Float16* Vtb = (_Float16*)take((size_t)MROWS * NE * 2);
  _Float16* ctx = (_Float16*)take((size_t)MROWS * NE * 2);
  (void)ws_size; (void)in_sizes; (void)n_in; (void)out_size;

  const int NELEM = MROWS * NE;             // 8M
  k_cvt_f16<<<NELEM / 256, 256, 0, stream>>>(x, xh, NELEM);
  k_w_transpose<<<(NE * NE) / 256, 256, 0, stream>>>(Wq, Wqt);
  k_w_transpose<<<(NE * NE) / 256, 256, 0, stream>>>(Wk, Wkt);
  k_w_transpose<<<(NE * NE) / 256, 256, 0, stream>>>(Wv, Wvt);
  k_w_transpose<<<(NE * NE) / 256, 256, 0, stream>>>(Wo, Wot);

  dim3 gg(MROWS / 64, NE / 64);
  k_gemm<<<gg, 128, 0, stream>>>(xh, Wqt, Y, nullptr, nullptr, MROWS, NE, NE);
  k_reshape_qk<<<NELEM / 256, 256, 0, stream>>>(Y, Qh, 0.125f);   // fold d^-1/2 into Q
  k_gemm<<<gg, 128, 0, stream>>>(xh, Wkt, Y, nullptr, nullptr, MROWS, NE, NE);
  k_reshape_qk<<<NELEM / 256, 256, 0, stream>>>(Y, Khb, 1.0f);
  k_gemm<<<gg, 128, 0, stream>>>(xh, Wvt, Y, nullptr, nullptr, MROWS, NE, NE);
  k_reshape_vt<<<NELEM / 256, 256, 0, stream>>>(Y, Vtb);

  k_flash<<<dim3(SEQ / 64, NH, NB), 128, 0, stream>>>(Qh, Khb, Vtb, ctx);

  k_gemm<<<gg, 128, 0, stream>>>(ctx, Wot, nullptr, out, bo, MROWS, NE, NE);
}
